// AttentionMatrix_81003083203500
// MI455X (gfx1250) — matrix operand order repaired and checked by reading, not compiled, not run on hardware
//
#include <hip/hip_runtime.h>
#include <math.h>

typedef __attribute__((ext_vector_type(16))) __bf16 v16bf;
typedef __attribute__((ext_vector_type(8)))  float  v8f;
typedef __attribute__((ext_vector_type(4)))  unsigned int u32x4;

namespace {
constexpr int kSeq   = 2048;
constexpr int kDim   = 512;
constexpr int kMTile = 64;    // q rows per workgroup
constexpr int kQS    = 520;   // padded LDS row stride in shorts (1040 B -> conflict-free)
constexpr int kNRows = 32;    // K rows staged per n-step (two 16-col tiles)

union Frag { v16bf v; u32x4 q[2]; };

__device__ __forceinline__ unsigned short bf16_rne(float f) {
  unsigned int u = __float_as_uint(f);
  u += 0x7fffu + ((u >> 16) & 1u);
  return (unsigned short)(u >> 16);
}
__device__ __forceinline__ float bf16_f32(unsigned short h) {
  return __uint_as_float((unsigned int)h << 16);
}
__device__ __forceinline__ void cvt_hi_lo4(unsigned short* hp, unsigned short* lp, float4 x) {
  ushort4 h, l;
  h.x = bf16_rne(x.x); l.x = bf16_rne(x.x - bf16_f32(h.x));
  h.y = bf16_rne(x.y); l.y = bf16_rne(x.y - bf16_f32(h.y));
  h.z = bf16_rne(x.z); l.z = bf16_rne(x.z - bf16_f32(h.z));
  h.w = bf16_rne(x.w); l.w = bf16_rne(x.w - bf16_f32(h.w));
  *reinterpret_cast<ushort4*>(hp) = h;
  *reinterpret_cast<ushort4*>(lp) = l;
}
} // namespace

__global__ __launch_bounds__(256)
void attn_causal_softmax_gfx1250(const float* __restrict__ Kin,
                                 const float* __restrict__ Qin,
                                 float* __restrict__ Out) {
  extern __shared__ unsigned char smem[];
  unsigned short* QH = reinterpret_cast<unsigned short*>(smem);
  unsigned short* QL = QH + kMTile * kQS;
  unsigned short* KH = QL + kMTile * kQS;
  unsigned short* KL = KH + kNRows * kQS;
  float* redM = reinterpret_cast<float*>(KL + kNRows * kQS);  // [8 waves][16 rows]
  float* redS = redM + 8 * 16;

  const int qb   = (gridDim.x - 1) - blockIdx.x;  // biggest causal blocks first
  const int b    = blockIdx.y;
  const int tid  = threadIdx.x;
  const int lane = tid & 31;
  const int wave = tid >> 5;
  const int qi   = wave >> 1;   // q-subtile 0..3
  const int ni   = wave & 1;    // n parity 0..1
  const int lm   = lane & 15;
  const int hh   = lane >> 4;
  const int qbase = qb * kMTile;
  const int rq    = qbase + qi * 16;  // wave's first global q row
  const int diagT = rq >> 4;          // last n-tile this wave must compute
  const float kScale  = 0.022097086912079608f;  // 1/sqrt(2048)
  const float kNegInf = -__builtin_huge_valf();

  const float* Kb = Kin + (size_t)b * kSeq * kDim;
  const float* Qb = Qin + (size_t)b * kSeq * kDim;
  float*       Ob = Out + (size_t)b * kSeq * kSeq;

  // ---- stage Q block once: fp32 -> bf16 hi + residual lo ----
  {
    const float* src = Qb + (size_t)qbase * kDim;
#pragma unroll 4
    for (int i = 0; i < (kMTile * kDim / 4) / 256; ++i) {
      int f = tid + 256 * i;
      int r = f >> 7;
      int c = (f & 127) << 2;
      float4 x = *reinterpret_cast<const float4*>(src + (size_t)r * kDim + c);
      cvt_hi_lo4(&QH[r * kQS + c], &QL[r * kQS + c], x);
    }
  }

  const unsigned short* qh = QH + (qi * 16 + lm) * kQS;
  const unsigned short* ql = QL + (qi * 16 + lm) * kQS;
  const unsigned short* kh = KH + (ni * 16 + lm) * kQS;
  const unsigned short* kl = KL + (ni * 16 + lm) * kQS;

  auto stage_k = [&](int st) {
    const float* src = Kb + (size_t)(kNRows * st) * kDim;
#pragma unroll 4
    for (int i = 0; i < (kNRows * kDim / 4) / 256; ++i) {
      int f = tid + 256 * i;
      int r = f >> 7;
      int c = (f & 127) << 2;
      float4 x = *reinterpret_cast<const float4*>(src + (size_t)r * kDim + c);
      cvt_hi_lo4(&KH[r * kQS + c], &KL[r * kQS + c], x);
    }
  };

  // 16x16 tile of Q.K^T over D=512, bf16 hi/lo split: hi*hi + hi*lo + lo*hi
  auto tile_qkT = [&]() -> v8f {
    v8f acc = {};
#pragma unroll
    for (int s = 0; s < 16; ++s) {
      const int k0 = 32 * s;
      Frag ah, al, bh, bl;
      ah.q[0] = *reinterpret_cast<const u32x4*>(qh + k0 + 8 * hh);
      ah.q[1] = *reinterpret_cast<const u32x4*>(qh + k0 + 16 + 8 * hh);
      al.q[0] = *reinterpret_cast<const u32x4*>(ql + k0 + 8 * hh);
      al.q[1] = *reinterpret_cast<const u32x4*>(ql + k0 + 16 + 8 * hh);
      bh.q[0] = *reinterpret_cast<const u32x4*>(kh + k0 + 8 * hh);
      bh.q[1] = *reinterpret_cast<const u32x4*>(kh + k0 + 16 + 8 * hh);
      bl.q[0] = *reinterpret_cast<const u32x4*>(kl + k0 + 8 * hh);
      bl.q[1] = *reinterpret_cast<const u32x4*>(kl + k0 + 16 + 8 * hh);
      acc = __builtin_amdgcn_wmma_f32_16x16x32_bf16(false, ah.v, false, bh.v,
                                                    (short)0, acc, false, false);
      acc = __builtin_amdgcn_wmma_f32_16x16x32_bf16(false, ah.v, false, bl.v,
                                                    (short)0, acc, false, false);
      acc = __builtin_amdgcn_wmma_f32_16x16x32_bf16(false, al.v, false, bh.v,
                                                    (short)0, acc, false, false);
    }
    return acc;
  };

  // online per-row state; acc slot v <-> global row rq + v + 8*hh
  float mrow[8], srow[8];
#pragma unroll
  for (int v = 0; v < 8; ++v) { mrow[v] = kNegInf; srow[v] = 0.0f; }

  const int nsteps = 2 * qb + 2;

  // ================= sweep 1: row max / sum(exp) =================
  for (int st = 0; st < nsteps; ++st) {
    __syncthreads();
    stage_k(st);
    __syncthreads();
    const int nt = 2 * st + ni;
    if (nt > diagT) continue;
    v8f acc = tile_qkT();
    const int cg = nt * 16 + lm;
#pragma unroll
    for (int v = 0; v < 8; ++v) {
      const int R = rq + v + 8 * hh;
      float x = (cg <= R) ? acc[v] * kScale : kNegInf;
      float tm = x;  // butterfly max across the 16 lanes of this half
      tm = fmaxf(tm, __shfl_xor(tm, 1, 32));
      tm = fmaxf(tm, __shfl_xor(tm, 2, 32));
      tm = fmaxf(tm, __shfl_xor(tm, 4, 32));
      tm = fmaxf(tm, __shfl_xor(tm, 8, 32));
      const float mnew = fmaxf(mrow[v], tm);  // tm finite: col c0 <= rq <= R
      float e = __expf(x - mnew);
      e += __shfl_xor(e, 1, 32);
      e += __shfl_xor(e, 2, 32);
      e += __shfl_xor(e, 4, 32);
      e += __shfl_xor(e, 8, 32);
      srow[v] = srow[v] * __expf(mrow[v] - mnew) + e;
      mrow[v] = mnew;
    }
  }

  // combine the two n-parity waves that share each row
  __syncthreads();
  if (lm == 0) {
#pragma unroll
    for (int v = 0; v < 8; ++v) {
      redM[(qi * 2 + ni) * 16 + v + 8 * hh] = mrow[v];
      redS[(qi * 2 + ni) * 16 + v + 8 * hh] = srow[v];
    }
  }
  __syncthreads();
  float mfin[8], sinv[8];
#pragma unroll
  for (int v = 0; v < 8; ++v) {
    const int r = v + 8 * hh;
    const float ma = redM[(qi * 2 + 0) * 16 + r];
    const float mb = redM[(qi * 2 + 1) * 16 + r];
    const float sa = redS[(qi * 2 + 0) * 16 + r];
    const float sb = redS[(qi * 2 + 1) * 16 + r];
    const float M = fmaxf(ma, mb);
    const float S = sa * __expf(ma - M) + sb * __expf(mb - M);
    mfin[v] = M;
    sinv[v] = 1.0f / S;  // S >= 1 (diagonal term), never 0
  }

  // ================= sweep 2: emit normalized probabilities =================
  for (int st = 0; st < nsteps; ++st) {
    __syncthreads();
    stage_k(st);
    __syncthreads();
    const int nt = 2 * st + ni;
    if (nt > diagT) continue;
    v8f acc = tile_qkT();
    const int cg = nt * 16 + lm;
#pragma unroll
    for (int v = 0; v < 8; ++v) {
      const int R = rq + v + 8 * hh;
      float x = (cg <= R) ? acc[v] * kScale : kNegInf;
      Ob[(size_t)R * kSeq + cg] = __expf(x - mfin[v]) * sinv[v];
    }
  }

  // ---- strictly-upper region past this wave's diagonal tile: pure zeros ----
  {
    const int cstart = rq + 16;  // cols >= here are zero for rows rq..rq+15
    const float4 z = make_float4(0.f, 0.f, 0.f, 0.f);
    for (int r = 0; r < 16; ++r) {
      float* rowp = Ob + (size_t)(rq + r) * kSeq;
      for (int c = cstart + 4 * (ni * 32 + lane); c < kSeq; c += 4 * 64) {
        *reinterpret_cast<float4*>(rowp + c) = z;
      }
    }
  }
}

extern "C" void kernel_launch(void* const* d_in, const int* in_sizes, int n_in,
                              void* d_out, int out_size, void* d_ws, size_t ws_size,
                              hipStream_t stream) {
  (void)in_sizes; (void)n_in; (void)out_size; (void)d_ws; (void)ws_size;
  const float* K = (const float*)d_in[0];  // setup_inputs order: K first
  const float* Q = (const float*)d_in[1];
  float* out = (float*)d_out;

  constexpr size_t lds_bytes =
      (size_t)(2 * kMTile * kQS + 2 * kNRows * kQS) * sizeof(unsigned short) +
      2 * 8 * 16 * sizeof(float);  // = 200,704 B (< 320 KB/WGP)

  dim3 grid(kSeq / kMTile, 8);  // 32 q-blocks x 8 batches = 256 workgroups
  attn_causal_softmax_gfx1250<<<grid, dim3(256), lds_bytes, stream>>>(K, Q, out);
}
